// DPSNLayer_13657996002041
// MI455X (gfx1250) — matrix operand order repaired and checked by reading, not compiled, not run on hardware
//
#include <hip/hip_runtime.h>
#include <hip/hip_bf16.h>

#define D_MODEL   1024
#define NUM_SLOTS 4096
#define T_TOKENS  4096   // B*S = 4*1024
#define MIN_K     8
#define MAX_K     32

typedef __bf16 bf16;
typedef bf16  v16bf __attribute__((ext_vector_type(16)));
typedef bf16  v8bf  __attribute__((ext_vector_type(8)));
typedef float v8f   __attribute__((ext_vector_type(8)));

// ---------------------------------------------------------------------------
// CDNA5 async direct-to-LDS copy (GLOBAL_LOAD_ASYNC_TO_LDS_B128, ASYNCcnt),
// with synchronous staging fallback if the builtin is unavailable.
// Signature (from compiler diagnostic): (v4i AS1*, v4i AS3*, imm off, imm cpol)
// ---------------------------------------------------------------------------
#if __has_builtin(__builtin_amdgcn_global_load_async_to_lds_b128)
#define HAVE_ASYNC_LDS 1
#endif

typedef int v4i_vs __attribute__((vector_size(4 * sizeof(int))));
typedef __attribute__((address_space(1))) v4i_vs g_v4i;
typedef __attribute__((address_space(3))) v4i_vs l_v4i;

// copy 16 bytes (8 bf16) global -> LDS
__device__ __forceinline__ void cp16(const bf16* g, bf16* l) {
#ifdef HAVE_ASYNC_LDS
  __builtin_amdgcn_global_load_async_to_lds_b128(
      (g_v4i*)(uintptr_t)g,
      (l_v4i*)(unsigned)(uintptr_t)l,   // low 32 bits = LDS byte offset
      0, 0);
#else
  *(uint4*)l = *(const uint4*)g;
#endif
}

template <int N>
__device__ __forceinline__ void wait_async() {
#ifdef HAVE_ASYNC_LDS
#if __has_builtin(__builtin_amdgcn_s_wait_asynccnt)
  __builtin_amdgcn_s_wait_asynccnt(N);
#else
  asm volatile("s_wait_asynccnt %0" ::"i"(N) : "memory");
#endif
#endif
}

// ---------------------------------------------------------------------------
// Kernel 1: convert x -> bf16, w_router -> bf16 transposed (slot-major),
// and zero the atomic accumulators (deterministic per launch).
// ---------------------------------------------------------------------------
__global__ __launch_bounds__(256) void prep_kernel(
    const float* __restrict__ x, const float* __restrict__ wr,
    bf16* __restrict__ xb, bf16* __restrict__ wb,
    float* __restrict__ counts, int* __restrict__ touched,
    float* __restrict__ keepsum) {
  const int i = blockIdx.x * 256 + threadIdx.x;
  if (i < T_TOKENS * D_MODEL) xb[i] = (bf16)x[i];
  if (i < NUM_SLOTS * D_MODEL) {
    const int n = i >> 10;        // slot
    const int d = i & (D_MODEL - 1);
    wb[i] = (bf16)wr[(size_t)d * NUM_SLOTS + n];   // transpose: [N][D]
  }
  if (i < NUM_SLOTS) { counts[i] = 0.0f; touched[i] = 0; }
  if (i == 0) *keepsum = 0.0f;
}

// ---------------------------------------------------------------------------
// Kernel 2: router GEMM, logits[T,N] = Xbf16 @ Wbf16^T via WMMA bf16.
// 128x128 tile per WG; 8 waves arranged 4(m) x 2(n); wave tile = 32x64.
// Double-buffered LDS, async direct-to-LDS staging pipelined against WMMA.
// ---------------------------------------------------------------------------
__global__ __launch_bounds__(256) void router_gemm(
    const bf16* __restrict__ xb, const bf16* __restrict__ wb,
    float* __restrict__ logits) {
  __shared__ __align__(64) bf16 ldsA[2][128 * 32];   // [buf][row][k] 8 KB each
  __shared__ __align__(64) bf16 ldsB[2][128 * 32];   // [buf][slot][k]

  const int tid  = threadIdx.x;
  const int lane = tid & 31;
  const int wave = tid >> 5;
  const int wm = wave & 3;          // 0..3 -> m offset 32*wm
  const int wn = wave >> 2;         // 0..1 -> n offset 64*wn
  const int row0 = blockIdx.x * 128;
  const int col0 = blockIdx.y * 128;

  v8f c[2][4];
  const v8f zero = {0.f,0.f,0.f,0.f,0.f,0.f,0.f,0.f};
#pragma unroll
  for (int mi = 0; mi < 2; ++mi)
#pragma unroll
    for (int ni = 0; ni < 4; ++ni) c[mi][ni] = zero;

  const int ldRow = tid >> 1;         // 0..127
  const int ldSeg = (tid & 1) * 16;   // bf16 elements: 0 or 16
  const bf16* gA = xb + (size_t)(row0 + ldRow) * D_MODEL + ldSeg;
  const bf16* gB = wb + (size_t)(col0 + ldRow) * D_MODEL + ldSeg;

  auto stage = [&](int buf, int kt) {
    const int k0 = kt * 32;
    bf16* dA = &ldsA[buf][ldRow * 32 + ldSeg];
    bf16* dB = &ldsB[buf][ldRow * 32 + ldSeg];
    cp16(gA + k0,     dA);
    cp16(gA + k0 + 8, dA + 8);
    cp16(gB + k0,     dB);
    cp16(gB + k0 + 8, dB + 8);
  };

  constexpr int NT = D_MODEL / 32;
  stage(0, 0);

  for (int kt = 0; kt < NT; ++kt) {
    const int buf = kt & 1;
    if (kt + 1 < NT) {
      stage(buf ^ 1, kt + 1);   // prefetch next tile into other buffer
      wait_async<4>();          // wait: this tile's 4 async loads done
#ifndef HAVE_ASYNC_LDS
      __builtin_prefetch(gA + (kt + 1) * 32, 0, 0);
      __builtin_prefetch(gB + (kt + 1) * 32, 0, 0);
#endif
    } else {
      wait_async<0>();
    }
    __syncthreads();

    // A fragment: lanes 0-15 hold M=0..15 K{0..7,16..23}; lanes 16-31 K{8..15,24..31}
    const int mrow  = lane & 15;
    const int khalf = lane >> 4;
    v16bf a[2], b[4];
#pragma unroll
    for (int mi = 0; mi < 2; ++mi) {
      const int r = wm * 32 + mi * 16 + mrow;
      ((v8bf*)&a[mi])[0] = *(const v8bf*)(&ldsA[buf][r * 32 + khalf * 8]);
      ((v8bf*)&a[mi])[1] = *(const v8bf*)(&ldsA[buf][r * 32 + 16 + khalf * 8]);
    }
    // B fragment: lane = column, lanes 0-15 K{0..7,16..23}; lanes 16-31 K{8..15,24..31}
#pragma unroll
    for (int ni = 0; ni < 4; ++ni) {
      const int r = wn * 64 + ni * 16 + (lane & 15);
      ((v8bf*)&b[ni])[0] = *(const v8bf*)(&ldsB[buf][r * 32 + khalf * 8]);
      ((v8bf*)&b[ni])[1] = *(const v8bf*)(&ldsB[buf][r * 32 + 16 + khalf * 8]);
    }
#pragma unroll
    for (int mi = 0; mi < 2; ++mi)
#pragma unroll
      for (int ni = 0; ni < 4; ++ni)
        c[mi][ni] = __builtin_amdgcn_wmma_f32_16x16x32_bf16(
            false, a[mi], false, b[ni], (short)0, c[mi][ni], false, false);
    __syncthreads();
  }

  // C/D layout: VGPR r, lanes 0-15: M=r,N=lane; lanes 16-31: M=8+r,N=lane-16
  const int mhalf = (lane >> 4) * 8;
  const int nl = lane & 15;
#pragma unroll
  for (int mi = 0; mi < 2; ++mi)
#pragma unroll
    for (int ni = 0; ni < 4; ++ni)
#pragma unroll
      for (int r = 0; r < 8; ++r) {
        const int m = row0 + wm * 32 + mi * 16 + mhalf + r;
        const int n = col0 + wn * 64 + ni * 16 + nl;
        logits[(size_t)m * NUM_SLOTS + n] = c[mi][ni][r];
      }
}

// ---------------------------------------------------------------------------
// Kernel 3: per-token softmax stats + exact top-32 (value desc, index asc),
// keep mask, normalized weights, scatter accumulators. One wave per token.
// ---------------------------------------------------------------------------
__global__ __launch_bounds__(256) void topk_kernel(
    const float* __restrict__ logits,
    float* __restrict__ rowmax, float* __restrict__ rowinv,
    int* __restrict__ top_idx, float* __restrict__ weights,
    float* __restrict__ counts, int* __restrict__ touched,
    float* __restrict__ keepsum) {
  __shared__ float lv[8][MAX_K];
  __shared__ int   li[8][MAX_K];
  const int lane = threadIdx.x & 31;
  const int w    = threadIdx.x >> 5;
  const int t    = blockIdx.x * 8 + w;
  const float* row = logits + (size_t)t * NUM_SLOTS;

  // row max
  float m = -1e30f;
  for (int i = lane; i < NUM_SLOTS; i += 32) m = fmaxf(m, row[i]);
#pragma unroll
  for (int off = 16; off; off >>= 1) m = fmaxf(m, __shfl_xor(m, off, 32));
  // sum exp
  float s = 0.0f;
  for (int i = lane; i < NUM_SLOTS; i += 32) s += __expf(row[i] - m);
#pragma unroll
  for (int off = 16; off; off >>= 1) s += __shfl_xor(s, off, 32);
  const float invZ = 1.0f / s;
  if (lane == 0) { rowmax[t] = m; rowinv[t] = invZ; }

  // threshold-descent top-32: pick strictly smaller (or same value, larger idx)
  float thrV = 3.0e38f; int thrI = -1;
  for (int k = 0; k < MAX_K; ++k) {
    float bv = -3.0e38f; int bi = 0x7fffffff;
    for (int i = lane; i < NUM_SLOTS; i += 32) {
      const float v = row[i];
      const bool cand = (v < thrV) || (v == thrV && i > thrI);
      if (cand && (v > bv || (v == bv && i < bi))) { bv = v; bi = i; }
    }
#pragma unroll
    for (int off = 16; off; off >>= 1) {
      const float ov = __shfl_xor(bv, off, 32);
      const int   oi = __shfl_xor(bi, off, 32);
      if (ov > bv || (ov == bv && oi < bi)) { bv = ov; bi = oi; }
    }
    thrV = bv; thrI = bi;
    if (lane == 0) { lv[w][k] = bv; li[w][k] = bi; }
  }
  __syncthreads();

  // lane k finishes entry k
  const float lval = lv[w][lane];
  const int   idx  = li[w][lane];
  const float prob = __expf(lval - m) * invZ;
  const float keepf = (lane < MIN_K || prob > (1.0f / NUM_SLOTS)) ? 1.0f : 0.0f;
  const float masked = prob * keepf;
  float msum = masked;
#pragma unroll
  for (int off = 16; off; off >>= 1) msum += __shfl_xor(msum, off, 32);
  weights[t * MAX_K + lane] = masked / (msum + 1e-9f);
  top_idx[t * MAX_K + lane] = idx;
  if (keepf > 0.0f) {
    atomicAdd(&counts[idx], 1.0f);
    atomicMax(&touched[idx], 1);
  }
  float kc = keepf;
#pragma unroll
  for (int off = 16; off; off >>= 1) kc += __shfl_xor(kc, off, 32);
  if (lane == 0) atomicAdd(keepsum, kc);
}

// ---------------------------------------------------------------------------
// Kernel 4: mean softmax prob per slot (coalesced column reduction, no atomics)
// ---------------------------------------------------------------------------
__global__ __launch_bounds__(256) void meanprobs_kernel(
    const float* __restrict__ logits, const float* __restrict__ rowmax,
    const float* __restrict__ rowinv, float* __restrict__ mean_probs) {
  const int n = blockIdx.x * 256 + threadIdx.x;
  float acc = 0.0f;
  for (int t = 0; t < T_TOKENS; ++t)
    acc += __expf(logits[(size_t)t * NUM_SLOTS + n] - rowmax[t]) * rowinv[t];
  mean_probs[n] = acc * (1.0f / T_TOKENS);
}

// ---------------------------------------------------------------------------
// Kernel 5: gather + proj + weighted recombine. One 256-thread WG per token.
// ---------------------------------------------------------------------------
__global__ __launch_bounds__(256) void output_kernel(
    const float* __restrict__ x, const float* __restrict__ pool,
    const int* __restrict__ top_idx, const float* __restrict__ weights,
    float* __restrict__ out) {
  __shared__ float red[8];
  const int t = blockIdx.x;
  const int tid = threadIdx.x, lane = tid & 31, wv = tid >> 5;
  float xr[4], acc[4] = {0.f, 0.f, 0.f, 0.f};
#pragma unroll
  for (int j = 0; j < 4; ++j) xr[j] = x[(size_t)t * D_MODEL + tid + 256 * j];

  for (int k = 0; k < MAX_K; ++k) {
    const float wgt = weights[t * MAX_K + k];
    const int   idx = top_idx[t * MAX_K + k];
    const float* p = pool + (size_t)idx * D_MODEL;
    float pv[4], partial = 0.0f;
#pragma unroll
    for (int j = 0; j < 4; ++j) { pv[j] = p[tid + 256 * j]; partial += xr[j] * pv[j]; }
#pragma unroll
    for (int off = 16; off; off >>= 1) partial += __shfl_xor(partial, off, 32);
    if (lane == 0) red[wv] = partial;
    __syncthreads();
    float dot = 0.0f;
#pragma unroll
    for (int i = 0; i < 8; ++i) dot += red[i];
    const float scale = wgt * dot;
#pragma unroll
    for (int j = 0; j < 4; ++j) acc[j] += scale * pv[j];
    __syncthreads();
  }
#pragma unroll
  for (int j = 0; j < 4; ++j) out[(size_t)t * D_MODEL + tid + 256 * j] = acc[j];
}

// ---------------------------------------------------------------------------
// Kernel 6: aux loss + active count
// ---------------------------------------------------------------------------
__global__ __launch_bounds__(256) void finalize_kernel(
    const float* __restrict__ mean_probs, const float* __restrict__ counts,
    const int* __restrict__ touched, const float* __restrict__ keepsum,
    float* __restrict__ out_scalars) {
  __shared__ float s1s[256], s2s[256];
  const int tid = threadIdx.x;
  float s1 = 0.0f, s2 = 0.0f;
  for (int n = tid; n < NUM_SLOTS; n += 256) {
    s1 += mean_probs[n] * counts[n];
    s2 += (float)touched[n];
  }
  s1s[tid] = s1; s2s[tid] = s2;
  __syncthreads();
  for (int off = 128; off; off >>= 1) {
    if (tid < off) { s1s[tid] += s1s[tid + off]; s2s[tid] += s2s[tid + off]; }
    __syncthreads();
  }
  if (tid == 0) {
    out_scalars[0] = (float)NUM_SLOTS * s1s[0] / (*keepsum + 1e-9f);
    out_scalars[1] = s2s[0];
  }
}

// ---------------------------------------------------------------------------
extern "C" void kernel_launch(void* const* d_in, const int* in_sizes, int n_in,
                              void* d_out, int out_size, void* d_ws, size_t ws_size,
                              hipStream_t stream) {
  const float* x    = (const float*)d_in[0];   // [4,1024,1024]
  const float* pool = (const float*)d_in[1];   // [4096,1024]
  const float* wr   = (const float*)d_in[2];   // [1024,4096]
  float* out = (float*)d_out;

  char* p = (char*)d_ws;
  bf16*  xb         = (bf16*)p;  p += (size_t)T_TOKENS * D_MODEL * 2;    // 8 MB
  bf16*  wb         = (bf16*)p;  p += (size_t)NUM_SLOTS * D_MODEL * 2;   // 8 MB
  float* logits     = (float*)p; p += (size_t)T_TOKENS * NUM_SLOTS * 4;  // 64 MB
  int*   top_idx    = (int*)p;   p += (size_t)T_TOKENS * MAX_K * 4;
  float* weights    = (float*)p; p += (size_t)T_TOKENS * MAX_K * 4;
  float* rowmax     = (float*)p; p += (size_t)T_TOKENS * 4;
  float* rowinv     = (float*)p; p += (size_t)T_TOKENS * 4;
  float* mean_probs = (float*)p; p += (size_t)NUM_SLOTS * 4;
  float* counts     = (float*)p; p += (size_t)NUM_SLOTS * 4;
  int*   touched    = (int*)p;   p += (size_t)NUM_SLOTS * 4;
  float* keepsum    = (float*)p; p += 256;

  prep_kernel<<<(T_TOKENS * D_MODEL + 255) / 256, 256, 0, stream>>>(
      x, wr, xb, wb, counts, touched, keepsum);

  dim3 ggrid(T_TOKENS / 128, NUM_SLOTS / 128);
  router_gemm<<<ggrid, 256, 0, stream>>>(xb, wb, logits);

  topk_kernel<<<T_TOKENS / 8, 256, 0, stream>>>(
      logits, rowmax, rowinv, top_idx, weights, counts, touched, keepsum);

  meanprobs_kernel<<<NUM_SLOTS / 256, 256, 0, stream>>>(
      logits, rowmax, rowinv, mean_probs);

  output_kernel<<<T_TOKENS, 256, 0, stream>>>(x, pool, top_idx, weights, out);

  finalize_kernel<<<1, 256, 0, stream>>>(
      mean_probs, counts, touched, keepsum, out + (size_t)T_TOKENS * D_MODEL);
}
